// MCInfoNCE_12962211299864
// MI455X (gfx1250) — matrix operand order repaired and checked by reading, not compiled, not run on hardware
//
#include <hip/hip_runtime.h>
#include <hip/hip_bf16.h>
#include <math.h>

typedef __attribute__((ext_vector_type(16))) _Float16 v16h;
typedef __attribute__((ext_vector_type(8)))  _Float16 v8h;
typedef __attribute__((ext_vector_type(8)))  float    v8f;

#define NSAMP   8
#define DIM     64
#define NNEG_C  128
#define WITERS  64
#define EPSV    1e-7f
#define BPAD    72   // padded halves per bbuf row (conflict-free, 16B aligned)
#define MPAD    68   // padded floats per mu_tile row (conflict-free, 16B aligned)

// ---------------- counter-based RNG (deterministic across replays) ----------------
__device__ __forceinline__ uint32_t lcg_hash(uint32_t x) {
  x ^= x >> 17; x *= 0xed5ad4bbu;
  x ^= x >> 11; x *= 0xac4c1b51u;
  x ^= x >> 15; x *= 0x31848babu;
  x ^= x >> 14; return x;
}
__device__ __forceinline__ uint32_t mix3(uint32_t a, uint32_t b, uint32_t c) {
  return lcg_hash(a * 0x9E3779B9u + b * 0x85EBCA77u + c * 0xC2B2AE3Du + 0x27D4EB2Fu);
}
__device__ __forceinline__ float u01(uint32_t h) {
  return ((float)(h >> 8) + 0.5f) * (1.0f / 16777216.0f);  // (0,1)
}
// Box-Muller pair m of a stream: normals for z-dims (2m, 2m+1)
__device__ __forceinline__ void bm_pair(uint32_t seed, int m, float& n0, float& n1) {
  uint32_t ha = lcg_hash(seed + (uint32_t)m * 0x68E31DA4u + 0xB5297A4Du);
  uint32_t hb = lcg_hash(seed ^ ((uint32_t)m * 0x1B56C4E9u + 0x9E3779B9u));
  float r = sqrtf(-2.0f * __logf(u01(ha)));
  float sn, cs;
  __sincosf(6.28318530718f * u01(hb), &sn, &cs);
  n0 = r * cs; n1 = r * sn;
}

// ---------------- wave32 reductions (used once per wave at the end) ----------------
__device__ __forceinline__ float wave_sum(float x) {
  #pragma unroll
  for (int o = 16; o > 0; o >>= 1) x += __shfl_xor(x, o, 32);
  return x;
}
__device__ __forceinline__ float wave_max(float x) {
  #pragma unroll
  for (int o = 16; o > 0; o >>= 1) x = fmaxf(x, __shfl_xor(x, o, 32));
  return x;
}

// ---------------- Wood's rejection sampler for vMF w (m = D-1 = 63) ----------------
__device__ __forceinline__ float sample_w(float kappa, uint32_t seed) {
  const float m = 63.0f;
  float sq = sqrtf(4.0f * kappa * kappa + m * m);
  float bb = (sq - 2.0f * kappa) / m;
  float aa = (m + 2.0f * kappa + sq) * 0.25f;
  float dd = 4.0f * aa * bb / (1.0f + bb) - m * __logf(m);
  float w = 1.0f;
  for (int it = 0; it < WITERS; ++it) {
    uint32_t h1 = lcg_hash(seed + (uint32_t)it * 0x6C8E9CF5u + 0x2545F491u);
    uint32_t h2 = lcg_hash(seed ^ ((uint32_t)it * 0x94D049BBu + 0x63641362u));
    float eps = fminf(fmaxf(u01(h1), EPSV), 1.0f - EPSV);
    float u   = u01(h2);
    float denom = 1.0f - (1.0f - bb) * eps;
    float cost  = (1.0f - (1.0f + bb) * eps) / denom;
    float t     = 2.0f * aa * bb / denom;
    if (m * __logf(t) - t + dd >= __logf(u)) { w = cost; break; }
  }
  return w;
}

// ---------------- lane-serial vMF sample: no cross-lane ops ----------------
// z = [w, sqrt(1-w^2) v/||v||]; Householder u = (e1-mu)/clip(||e1-mu||,EPS);
// out = z - 2 (z.u) u.  ||e1-mu||^2 = 2-2*mu[0] analytically (mu unit).
// Pass 1 accumulates Sum g^2 and Sum g*mu in registers; pass 2 regenerates the
// same normals (counter RNG) and writes packed f16 pairs.
__device__ __forceinline__ void gen_sample_serial(
    const float* __restrict__ mu, float w, uint32_t seed,
    _Float16* __restrict__ dst,                 // 64 contiguous halves
    const _Float16* __restrict__ dotbuf,        // optional: f16 row to dot against
    float* __restrict__ dot_out) {
  const float mu0 = mu[0];
  float gg = 0.0f, gm = 0.0f;
  #pragma unroll 4
  for (int m = 0; m < 32; ++m) {
    float n0, n1;
    bm_pair(seed, m, n0, n1);
    if (m > 0) { gg += n0 * n0; gm += n0 * mu[2 * m]; }
    gg += n1 * n1; gm += n1 * mu[2 * m + 1];
  }
  float scale = sqrtf(fmaxf(1.0f - w * w, EPSV)) * rsqrtf(fmaxf(gg, 1e-30f));
  float un    = fmaxf(sqrtf(fmaxf(2.0f - 2.0f * mu0, 0.0f)), EPSV);
  float dzm   = w * mu0 + scale * gm;         // z . mu
  float c     = 2.0f * ((w - dzm) / un) / un; // 2 (z.u_hat)/||u_vec||
  float dacc  = 0.0f;
  #pragma unroll 4
  for (int m = 0; m < 32; ++m) {
    float n0, n1;
    bm_pair(seed, m, n0, n1);
    float oe = (m == 0) ? (w - c * (1.0f - mu0)) : (scale * n0 + c * mu[2 * m]);
    float oo = scale * n1 + c * mu[2 * m + 1];
    if (dotbuf) dacc += oe * (float)dotbuf[2 * m] + oo * (float)dotbuf[2 * m + 1];
    union { _Float16 h[2]; uint32_t u; } pk;
    pk.h[0] = (_Float16)oe; pk.h[1] = (_Float16)oo;
    *(uint32_t*)(dst + 2 * m) = pk.u;         // packed b32 store
  }
  if (dot_out) *dot_out = dacc;
}

// ---------------- WMMA fragment loaders (CDNA5 ISA layouts) ----------------
__device__ __forceinline__ v16h load_a_frag(const _Float16* __restrict__ a, int lane, int c) {
  int M  = lane & 15;
  int k0 = c * 32 + ((lane < 16) ? 0 : 8);
  v8h lo = *(const v8h*)(a + M * DIM + k0);
  v8h hi = *(const v8h*)(a + M * DIM + k0 + 16);
  return __builtin_shufflevector(lo, hi, 0,1,2,3,4,5,6,7,8,9,10,11,12,13,14,15);
}
__device__ __forceinline__ v16h load_b_frag(const _Float16* __restrict__ bmat, int lane, int c) {
  int N  = lane & 15;
  int k0 = c * 32 + ((lane < 16) ? 0 : 8);
  const _Float16* p = bmat + N * BPAD + k0;   // 16B-aligned pieces
  v8h lo = *(const v8h*)p;
  v8h hi = *(const v8h*)(p + 16);
  return __builtin_shufflevector(lo, hi, 0,1,2,3,4,5,6,7,8,9,10,11,12,13,14,15);
}

__global__ void zero_out_kernel(float* out) { out[0] = 0.0f; }

__global__ __launch_bounds__(256) void mcinfonce_kernel(
    const float* __restrict__ mu_ref, const float* __restrict__ kappa_ref,
    const float* __restrict__ mu_pos, const float* __restrict__ kappa_pos,
    const float* __restrict__ mu_neg, const float* __restrict__ kappa_neg,
    const float* __restrict__ loss_kappa, float* __restrict__ out, int Bsz) {
  __shared__ __attribute__((aligned(16))) _Float16 aref[16 * DIM];          // 2 KB
  __shared__ __attribute__((aligned(16))) _Float16 bbuf[8][2][16 * BPAD];   // 36 KB
  __shared__ __attribute__((aligned(16))) float    mu_tile[8][16][MPAD];    // 34.8 KB
  __shared__ __attribute__((aligned(16))) _Float16 posbuf[NSAMP][DIM];      // 1 KB
  __shared__ __attribute__((aligned(16))) float    sims[NSAMP][NNEG_C];     // 4 KB
  __shared__ float sp_l[NSAMP];
  __shared__ float term_l[NSAMP];

  const int b    = blockIdx.x;
  const int tid  = threadIdx.x;
  const int wv   = tid >> 5;
  const int lane = tid & 31;
  const float lk = loss_kappa[0];

  // ---- stage this wave's 16 mu_neg rows into padded LDS (coalesced float4) ----
  for (int i = lane; i < 256; i += 32) {
    int rr = i >> 4, cc = (i & 15) * 4;
    float4 v = *(const float4*)(mu_neg + ((size_t)b * NNEG_C + wv * 16 + rr) * DIM + cc);
    *(float4*)&mu_tile[wv][rr][cc] = v;
  }

  // ---- ref & pos samples: lane-serial on wave 0, lanes 0..7 ----
  if (wv == 0 && lane < NSAMP) {
    const int s = lane;
    float w_r = sample_w(kappa_ref[b], mix3(0xA11u, (uint32_t)b, (uint32_t)s));
    gen_sample_serial(mu_ref + (size_t)b * DIM, w_r,
                      mix3(0xA33u, (uint32_t)b, (uint32_t)s),
                      aref + s * DIM, (const _Float16*)0, (float*)0);
    float w_p = sample_w(kappa_pos[b], mix3(0xB22u, (uint32_t)b, (uint32_t)s));
    float dot;
    gen_sample_serial(mu_pos + (size_t)b * DIM, w_p,
                      mix3(0xB44u, (uint32_t)b, (uint32_t)s),
                      posbuf[s], aref + s * DIM, &dot);
    sp_l[s] = lk * dot;
  }
  if (wv == 1) {  // zero A rows 8..15
    v8h z = {};
    *(v8h*)&aref[8 * DIM + lane * 16]     = z;
    *(v8h*)&aref[8 * DIM + lane * 16 + 8] = z;
  }
  __syncthreads();

  const v16h afrag0 = load_a_frag(aref, lane, 0);
  const v16h afrag1 = load_a_frag(aref, lane, 1);

  // ---- negatives: wave wv owns negs 16wv..16wv+15; round r covers s=2r,2r+1 ----
  const int   j    = lane & 15;
  const int   half = (lane >> 4) & 1;
  const int   n    = wv * 16 + j;
  const float kap  = kappa_neg[(size_t)b * NNEG_C + n];
  const float* murow = &mu_tile[wv][j][0];

  for (int r = 0; r < 4; ++r) {
    const int s0 = 2 * r + half;  // sample index this lane generates
    float wn = sample_w(kap, mix3(0xC55u + (uint32_t)s0 * 0x51u, (uint32_t)b, (uint32_t)n));
    gen_sample_serial(murow, wn,
                      mix3(0xC66u + (uint32_t)s0, (uint32_t)b, (uint32_t)n),
                      &bbuf[wv][half][j * BPAD], (const _Float16*)0, (float*)0);
    __builtin_amdgcn_wave_barrier();  // LDS stores precede fragment loads (in-wave)

    #pragma unroll
    for (int t = 0; t < 2; ++t) {
      const int s = 2 * r + t;
      v16h b0 = load_b_frag(&bbuf[wv][t][0], lane, 0);
      v16h b1 = load_b_frag(&bbuf[wv][t][0], lane, 1);
      v8f acc = {};
      acc = __builtin_amdgcn_wmma_f32_16x16x32_f16(false, afrag0, false, b0, (short)0, acc, false, false);
      acc = __builtin_amdgcn_wmma_f32_16x16x32_f16(false, afrag1, false, b1, (short)0, acc, false, false);
      // row s of D: VGPR s, lanes 0..15 (col = lane) -> store, reduce later
      float rsel = 0.0f;
      #pragma unroll
      for (int k = 0; k < 8; ++k) rsel = (s == k) ? acc[k] : rsel;
      if (lane < 16) sims[s][wv * 16 + lane] = rsel * lk;
    }
  }
  __syncthreads();

  // ---- per-sample LSE over 128 negatives: wave wv reduces sample s = wv ----
  {
    const int s = wv;
    float x0 = sims[s][lane];
    float x1 = sims[s][lane + 32];
    float x2 = sims[s][lane + 64];
    float x3 = sims[s][lane + 96];
    float mx = fmaxf(fmaxf(x0, x1), fmaxf(x2, x3));
    mx = wave_max(mx);
    float se = __expf(x0 - mx) + __expf(x1 - mx) + __expf(x2 - mx) + __expf(x3 - mx);
    se = wave_sum(se);
    if (lane == 0) {
      float neg_lse = mx + __logf(se);
      float a  = sp_l[s];
      float mm = fmaxf(a, neg_lse);
      float ld = mm + __logf(__expf(a - mm) + __expf(neg_lse - mm));
      term_l[s] = a - ld;
    }
  }
  __syncthreads();

  // ---- final combine for this batch element ----
  if (tid == 0) {
    float tm = -3.4e38f;
    #pragma unroll
    for (int s = 0; s < NSAMP; ++s) tm = fmaxf(tm, term_l[s]);
    float ts = 0.0f;
    #pragma unroll
    for (int s = 0; s < NSAMP; ++s) ts += __expf(term_l[s] - tm);
    float logprob = tm + __logf(ts) - __logf((float)NSAMP);
    atomicAdd(out, -logprob / (float)Bsz);
  }
}

extern "C" void kernel_launch(void* const* d_in, const int* in_sizes, int n_in,
                              void* d_out, int out_size, void* d_ws, size_t ws_size,
                              hipStream_t stream) {
  const float* mu_ref     = (const float*)d_in[0];
  const float* kappa_ref  = (const float*)d_in[1];
  const float* mu_pos     = (const float*)d_in[2];
  const float* kappa_pos  = (const float*)d_in[3];
  const float* mu_neg     = (const float*)d_in[4];
  const float* kappa_neg  = (const float*)d_in[5];
  const float* loss_kappa = (const float*)d_in[6];
  float* out = (float*)d_out;
  const int Bsz = in_sizes[1];  // batch size from kappa_ref

  zero_out_kernel<<<1, 1, 0, stream>>>(out);
  mcinfonce_kernel<<<Bsz, 256, 0, stream>>>(mu_ref, kappa_ref, mu_pos, kappa_pos,
                                            mu_neg, kappa_neg, loss_kappa, out, Bsz);
}
